// ReverseProcess_29377576304940
// MI455X (gfx1250) — compile-verified, matrix operand order checked by reading, not run on hardware
//
#include <hip/hip_runtime.h>
#include <hip/hip_bf16.h>
#include <math.h>

typedef __attribute__((ext_vector_type(16))) _Float16 v16h;
typedef __attribute__((ext_vector_type(8)))  float    v8f;

// ---------------------------------------------------------------------------
// threefry2x32 (JAX-compatible, 20 rounds)
// ---------------------------------------------------------------------------
__device__ __forceinline__ unsigned rotl32(unsigned v, int r) {
  return (v << r) | (v >> (32 - r));
}

__device__ __forceinline__ void threefry2x32(unsigned k0, unsigned k1,
                                             unsigned x0, unsigned x1,
                                             unsigned& o0, unsigned& o1) {
  unsigned ks2 = 0x1BD11BDAu ^ k0 ^ k1;
  x0 += k0; x1 += k1;
#define TF_G(a, b, c, d)                                   \
  x0 += x1; x1 = rotl32(x1, a); x1 ^= x0;                  \
  x0 += x1; x1 = rotl32(x1, b); x1 ^= x0;                  \
  x0 += x1; x1 = rotl32(x1, c); x1 ^= x0;                  \
  x0 += x1; x1 = rotl32(x1, d); x1 ^= x0;
  TF_G(13, 15, 26, 6)   x0 += k1;  x1 += ks2 + 1u;
  TF_G(17, 29, 16, 24)  x0 += ks2; x1 += k0 + 2u;
  TF_G(13, 15, 26, 6)   x0 += k0;  x1 += k1 + 3u;
  TF_G(17, 29, 16, 24)  x0 += k1;  x1 += ks2 + 4u;
  TF_G(13, 15, 26, 6)   x0 += ks2; x1 += k0 + 5u;
#undef TF_G
  o0 = x0; o1 = x1;
}

// Branch-free single-precision erfinv (Giles 2012 — same coefficients as
// XLA's ErfInv32, i.e. what the JAX reference actually evaluates).
__device__ __forceinline__ float erfinv_bf(float x) {
  float w = -__logf((1.0f - x) * (1.0f + x));
  bool  c = (w < 5.0f);
  float ww = c ? (w - 2.5f) : (__builtin_amdgcn_sqrtf(w) - 3.0f);
  float p;
  p =          c ?  2.81022636e-08f : -0.000200214257f;
  p = fmaf(p, ww, c ?  3.43273939e-07f :  0.000100950558f);
  p = fmaf(p, ww, c ? -3.5233877e-06f  :  0.00134934322f);
  p = fmaf(p, ww, c ? -4.39150654e-06f : -0.00367342844f);
  p = fmaf(p, ww, c ?  0.00021858087f  :  0.00573950773f);
  p = fmaf(p, ww, c ? -0.00125372503f  : -0.0076224613f);
  p = fmaf(p, ww, c ? -0.00417768164f  :  0.00943887047f);
  p = fmaf(p, ww, c ?  0.246640727f    :  1.00167406f);
  p = fmaf(p, ww, c ?  1.50140941f     :  2.83297682f);
  return p * x;
}

// bits -> uniform(-0.99999994, 1) -> sqrt(2)*erfinv(u)  (matches jax.random.normal)
__device__ __forceinline__ float bits_to_normal(unsigned bits) {
  unsigned ub = (bits >> 9) | 0x3f800000u;
  float f = __uint_as_float(ub) - 1.0f;        // [0,1)
  const float lo = -0.99999994f;               // nextafter(-1, 0)
  float u = f * (1.0f - lo) + lo;
  u = fmaxf(u, lo);
  return 1.4142135623730951f * erfinv_bf(u);
}

// ---------------------------------------------------------------------------
// Prologue: per-t records of 8 dwords at ws[t*8]:
//   [0]=coefA  [1]=invSqrtAlpha  [2]=sigma  [3]=temb
//   [4]=fk0    [5]=fk1 (folded threefry key, as u32)   [6,7]=pad
// ---------------------------------------------------------------------------
__global__ void precompute_kernel(const float* __restrict__ betas,
                                  const float* __restrict__ tew,
                                  const float* __restrict__ teb,
                                  const int* __restrict__ n_steps_p,
                                  float* __restrict__ ws) {
  if (threadIdx.x != 0 || blockIdx.x != 0) return;
  int T = n_steps_p[0];

  // zero record 0 (read harmlessly by the pipelined prefetch at t==1)
  for (int q = 0; q < 8; ++q) ws[q] = 0.0f;

  float abar_km1 = 1.0f;  // alpha_bar[k-1]  (abar[-1] := 1)
  for (int k = 0; k <= T && k < 256; ++k) {
    float alpha = 1.0f - betas[k];
    float abar_k = abar_km1 * alpha;
    if (k >= 1) {
      float* rec = ws + (size_t)k * 8;
      rec[0] = (1.0f - alpha) * rsqrtf(1.0f - abar_k);
      rec[1] = rsqrtf(alpha);
      rec[2] = (k == 1) ? 0.0f
                        : sqrtf((1.0f - alpha) * (1.0f - abar_km1) / (1.0f - abar_k));
      rec[3] = tew[k - 1] + teb[0];
      unsigned o0, o1;
      threefry2x32(0u, 1u, 0u, (unsigned)k, o0, o1);  // fold_in(key(1), t)
      ((unsigned*)rec)[4] = o0;
      ((unsigned*)rec)[5] = o1;
      rec[6] = 0.0f; rec[7] = 0.0f;
    }
    abar_km1 = abar_k;
  }
}

// ---------------------------------------------------------------------------
// WMMA helpers
// ---------------------------------------------------------------------------
__device__ __forceinline__ v8f wmma16(v16h a, v16h b, v8f c) {
  return __builtin_amdgcn_wmma_f32_16x16x32_f16(false, a, false, b, (short)0, c,
                                                false, false);
}

// Build f16 A-fragment (16x32 tile of weight matrix W[O][I], row-major):
// A(m,k_local) = W[ob+m][kb + k_local], zero-padded.
// 16-bit A layout: lane m = lane&15; lanes<16 hold K = {0..7, 16..23},
// lanes>=16 hold K = {8..15, 24..31}.
__device__ __forceinline__ v16h build_a(const float* __restrict__ W,
                                        int O, int I, int ob, int kb, int lane) {
  int m = lane & 15;
  int koff = (lane & 16) ? 8 : 0;
  v16h a;
#pragma unroll
  for (int j = 0; j < 16; ++j) {
    int k = kb + koff + j + ((j & 8) ? 8 : 0);  // j<8: K=koff+j ; j>=8: K=16+koff+(j-8)
    float v = 0.0f;
    if ((ob + m) < O && k < I) v = W[(ob + m) * I + k];
    a[j] = (_Float16)v;
  }
  return a;
}

// Pack two f32 D tiles (features 0..15 in `lo`, 16..31 in `hi`) into the next
// layer's 16-bit B fragment. Lane-aligned: B[e]=lo[e], B[8+e]=hi[e].
__device__ __forceinline__ v16h pack_b(const v8f& lo, const v8f& hi) {
  v16h b;
#pragma unroll
  for (int e = 0; e < 8; ++e) {
    b[e]     = (_Float16)lo[e];
    b[8 + e] = (_Float16)hi[e];
  }
  return b;
}

__device__ __forceinline__ void relu8(v8f& d) {
#pragma unroll
  for (int r = 0; r < 8; ++r) d[r] = fmaxf(d[r], 0.0f);
}

// Raw threefry bits for step t (branch-free integer chain; t==0 reads the
// zeroed record and the result is discarded).
__device__ __forceinline__ unsigned bits_for(const float* __restrict__ ws, int t,
                                             unsigned ctr0, unsigned ctr1,
                                             bool hiw) {
  const unsigned* rec = (const unsigned*)(ws + (size_t)t * 8);
  unsigned r0, r1;
  threefry2x32(rec[4], rec[5], ctr0, ctr1, r0, r1);
  return hiw ? r1 : r0;
}

// All weight/bias fragments (shared by both batch chains).
struct Frags {
  v16h A1[2], A2[4], A3k0[2], A3k1[2], A4f;
  v8f  cb1[2], cb2[4], cb3[2], cb4;
};

// Two independent 4-layer MLP chains, explicitly interleaved so every WMMA is
// followed in issue order by an independent WMMA (disjoint registers) that
// legally fills its RAW-hazard shadow — no v_nops needed.
__device__ __forceinline__ void mlp_eval2(const Frags& F,
                                          const v16h& binA, const v16h& binB,
                                          v8f& outA, v8f& outB) {
  // layer 1: 3 -> 32
  v8f dA0 = wmma16(F.A1[0], binA, F.cb1[0]);
  v8f dB0 = wmma16(F.A1[0], binB, F.cb1[0]);
  v8f dA1 = wmma16(F.A1[1], binA, F.cb1[1]);
  v8f dB1 = wmma16(F.A1[1], binB, F.cb1[1]);
  relu8(dA0); relu8(dB0); relu8(dA1); relu8(dB1);
  v16h b2A = pack_b(dA0, dA1);
  v16h b2B = pack_b(dB0, dB1);

  // layer 2: 32 -> 64 (8 interleaved WMMAs)
  v8f d2A[4], d2B[4];
#pragma unroll
  for (int ot = 0; ot < 4; ++ot) {
    d2A[ot] = wmma16(F.A2[ot], b2A, F.cb2[ot]);
    d2B[ot] = wmma16(F.A2[ot], b2B, F.cb2[ot]);
  }
#pragma unroll
  for (int ot = 0; ot < 4; ++ot) { relu8(d2A[ot]); relu8(d2B[ot]); }
  v16h b3Aa = pack_b(d2A[0], d2A[1]);   // A: K = 0..31
  v16h b3Ab = pack_b(d2A[2], d2A[3]);   // A: K = 32..63
  v16h b3Ba = pack_b(d2B[0], d2B[1]);   // B: K = 0..31
  v16h b3Bb = pack_b(d2B[2], d2B[3]);   // B: K = 32..63

  // layer 3: 64 -> 32 (each accumulator's dependent 2nd WMMA is 3 independent
  // WMMAs away in issue order)
  v8f tA0 = wmma16(F.A3k0[0], b3Aa, F.cb3[0]);
  v8f tB0 = wmma16(F.A3k0[0], b3Ba, F.cb3[0]);
  v8f tA1 = wmma16(F.A3k0[1], b3Aa, F.cb3[1]);
  v8f tB1 = wmma16(F.A3k0[1], b3Ba, F.cb3[1]);
  tA0 = wmma16(F.A3k1[0], b3Ab, tA0);
  tB0 = wmma16(F.A3k1[0], b3Bb, tB0);
  tA1 = wmma16(F.A3k1[1], b3Ab, tA1);
  tB1 = wmma16(F.A3k1[1], b3Bb, tB1);
  relu8(tA0); relu8(tB0); relu8(tA1); relu8(tB1);
  v16h b4A = pack_b(tA0, tA1);
  v16h b4B = pack_b(tB0, tB1);

  // layer 4: 32 -> 2 (padded to 16 rows; bias in C)
  outA = wmma16(F.A4f, b4A, F.cb4);
  outB = wmma16(F.A4f, b4B, F.cb4);
}

// ---------------------------------------------------------------------------
// Main persistent kernel: each wave owns 32 elements as TWO independent
// batches of 16 (A: base+n, B: base+16+n). The two MLP chains are explicitly
// interleaved; noise for step t-1 (threefry) and the erfinv of the current
// bits overlap the WMMAs as well.
// ---------------------------------------------------------------------------
__global__ __launch_bounds__(256) void diffusion_kernel(
    const float* __restrict__ x_init,
    const float* __restrict__ w1, const float* __restrict__ b1,
    const float* __restrict__ w2, const float* __restrict__ b2,
    const float* __restrict__ w3, const float* __restrict__ b3,
    const float* __restrict__ w4, const float* __restrict__ b4,
    const int* __restrict__ n_steps_p, const float* __restrict__ ws,
    float* __restrict__ out, int N) {
  int tid  = blockIdx.x * blockDim.x + threadIdx.x;
  int lane = threadIdx.x & 31;
  int wave = tid >> 5;
  int n    = lane & 15;
  int base = wave * 32;
  int iA   = base + n;         // batch A element (mirrored across lane halves)
  int iB   = base + 16 + n;    // batch B element (mirrored across lane halves)
  int iiA  = (iA < N) ? iA : 0;
  int iiB  = (iB < N) ? iB : 0;

  // ---- shared weight/bias fragments ----
  Frags F;
#pragma unroll
  for (int ot = 0; ot < 2; ++ot) F.A1[ot] = build_a(w1, 32, 3, ot * 16, 0, lane);
#pragma unroll
  for (int ot = 0; ot < 4; ++ot) F.A2[ot] = build_a(w2, 64, 32, ot * 16, 0, lane);
#pragma unroll
  for (int ot = 0; ot < 2; ++ot) {
    F.A3k0[ot] = build_a(w3, 32, 64, ot * 16, 0, lane);
    F.A3k1[ot] = build_a(w3, 32, 64, ot * 16, 32, lane);
  }
  F.A4f = build_a(w4, 2, 32, 0, 0, lane);

  int hoff = (lane & 16) ? 8 : 0;
#pragma unroll
  for (int ot = 0; ot < 2; ++ot)
#pragma unroll
    for (int r = 0; r < 8; ++r) F.cb1[ot][r] = b1[ot * 16 + hoff + r];
#pragma unroll
  for (int ot = 0; ot < 4; ++ot)
#pragma unroll
    for (int r = 0; r < 8; ++r) F.cb2[ot][r] = b2[ot * 16 + hoff + r];
#pragma unroll
  for (int ot = 0; ot < 2; ++ot)
#pragma unroll
    for (int r = 0; r < 8; ++r) F.cb3[ot][r] = b3[ot * 16 + hoff + r];
#pragma unroll
  for (int r = 0; r < 8; ++r) F.cb4[r] = (r < 2) ? b4[r] : 0.0f;

  float x0a = x_init[iiA * 2 + 0], x1a = x_init[iiA * 2 + 1];
  float x0b = x_init[iiB * 2 + 0], x1b = x_init[iiB * 2 + 1];

  int T = n_steps_p[0];

  // ---- loop-invariant noise counters (per chain) ----
  // jax bits array: n_total = 2N, half = N. Lower lane half computes component
  // c=0 (j=2i), upper half c=1 (j=2i+1); results shared via shuffle.
  unsigned half = (unsigned)N;
  unsigned hsel = (lane & 16) ? 1u : 0u;
  unsigned jA = (((unsigned)iiA) << 1) | hsel;
  unsigned jB = (((unsigned)iiB) << 1) | hsel;
  bool     hiwA = (jA >= half), hiwB = (jB >= half);
  unsigned cA0 = hiwA ? (jA - half) : jA, cA1 = cA0 + half;
  unsigned cB0 = hiwB ? (jB - half) : jB, cB1 = cB0 + half;

  bool lohalf = (lane < 16);

  // prime the noise pipelines with step T's raw bits
  unsigned bitsA = bits_for(ws, T, cA0, cA1, hiwA);
  unsigned bitsB = bits_for(ws, T, cB0, cB1, hiwB);

  // input B fragments hoisted: elements 3..15 stay zero in loop-carried regs
  v16h binA = {}, binB = {};

  for (int t = T; t >= 1; --t) {
    const float* rec = ws + (size_t)t * 8;
    float ca = rec[0], ia = rec[1], sg = rec[2], te = rec[3];

    // this step's normals + next step's bits: independent chains that fill
    // the WMMA shadows
    float zlA = bits_to_normal(bitsA);
    float zlB = bits_to_normal(bitsB);
    unsigned bitsA_n = bits_for(ws, t - 1, cA0, cA1, hiwA);
    unsigned bitsB_n = bits_for(ws, t - 1, cB0, cB1, hiwB);

    _Float16 teh = (_Float16)te;
    binA[0] = lohalf ? (_Float16)x0a : (_Float16)0.0f;
    binA[1] = lohalf ? (_Float16)x1a : (_Float16)0.0f;
    binA[2] = lohalf ? teh : (_Float16)0.0f;
    binB[0] = lohalf ? (_Float16)x0b : (_Float16)0.0f;
    binB[1] = lohalf ? (_Float16)x1b : (_Float16)0.0f;
    binB[2] = lohalf ? teh : (_Float16)0.0f;

    // two explicitly interleaved MLP chains (22 WMMAs)
    v8f d4a, d4b;
    mlp_eval2(F, binA, binB, d4a, d4b);

    float e0a = __shfl(d4a[0], n, 32);
    float e1a = __shfl(d4a[1], n, 32);
    float e0b = __shfl(d4b[0], n, 32);
    float e1b = __shfl(d4b[1], n, 32);

    float z0a = __shfl(zlA, n, 32);
    float z1a = __shfl(zlA, n + 16, 32);
    float z0b = __shfl(zlB, n, 32);
    float z1b = __shfl(zlB, n + 16, 32);

    x0a = (x0a - e0a * ca) * ia + sg * z0a;
    x1a = (x1a - e1a * ca) * ia + sg * z1a;
    x0b = (x0b - e0b * ca) * ia + sg * z0b;
    x1b = (x1b - e1b * ca) * ia + sg * z1b;

    bitsA = bitsA_n;
    bitsB = bitsB_n;
  }

  // lane l stores element base+l: lower half batch A, upper half batch B
  int istore = lohalf ? iA : iB;
  float s0 = lohalf ? x0a : x0b;
  float s1 = lohalf ? x1a : x1b;
  if (istore < N) {
    out[istore * 2 + 0] = s0;
    out[istore * 2 + 1] = s1;
  }
}

// ---------------------------------------------------------------------------
extern "C" void kernel_launch(void* const* d_in, const int* in_sizes, int n_in,
                              void* d_out, int out_size, void* d_ws, size_t ws_size,
                              hipStream_t stream) {
  const float* x_init = (const float*)d_in[0];
  const float* betas  = (const float*)d_in[1];
  const float* tew    = (const float*)d_in[2];
  const float* teb    = (const float*)d_in[3];
  const float* w1     = (const float*)d_in[4];
  const float* b1     = (const float*)d_in[5];
  const float* w2     = (const float*)d_in[6];
  const float* b2     = (const float*)d_in[7];
  const float* w3     = (const float*)d_in[8];
  const float* b3     = (const float*)d_in[9];
  const float* w4     = (const float*)d_in[10];
  const float* b4     = (const float*)d_in[11];
  const int* n_steps  = (const int*)d_in[12];
  float* ws  = (float*)d_ws;
  float* out = (float*)d_out;

  int N = in_sizes[0] / 2;

  precompute_kernel<<<1, 32, 0, stream>>>(betas, tew, teb, n_steps, ws);

  int waves = (N + 31) / 32;
  long threads = (long)waves * 32;
  int block = 256;
  int grid = (int)((threads + block - 1) / block);
  diffusion_kernel<<<grid, block, 0, stream>>>(x_init, w1, b1, w2, b2, w3, b3,
                                               w4, b4, n_steps, ws, out, N);
}
